// CoAttentionModule_54082228191250
// MI455X (gfx1250) — matrix operand order repaired and checked by reading, not compiled, not run on hardware
//
#include <hip/hip_runtime.h>
#include <hip/hip_bf16.h>

// ---------------------------------------------------------------------------
// CDNA5 (gfx1250) co-attention. All matmuls lower to v_wmma_f32_16x16x32_bf16.
//
// B=4, C=2048, H=W=32, NUM_HEADS=8, HIDDEN=256; M = B*H*W = 4096.
// 8 GEMMs x 34.4 GFLOP dominate (~275 GFLOP); compute-bound at bf16-WMMA rates.
// This round: LDS staging via GLOBAL_LOAD_ASYNC_TO_LDS_B128 (ASYNCcnt) so the
// k+1 tile copy has no VGPR round trip and overlaps the k-tile WMMAs fully;
// drain with s_wait_asynccnt 0 before each barrier.
// ---------------------------------------------------------------------------

typedef __attribute__((ext_vector_type(16))) __bf16 v16bf;
typedef __attribute__((ext_vector_type(8)))  float  v8f;

union BF16x16 { v16bf v; uint4 q[2]; };

__device__ __forceinline__ v8f v8f_zero() {
  v8f z;
  for (int i = 0; i < 8; ++i) z[i] = 0.f;
  return z;
}

// Async memory->LDS copy, 16B per lane. dst must point into __shared__;
// flat LDS-aperture addresses keep the LDS offset in their low 32 bits (ISA 10.2), so the
// 32-bit truncation is the wave-relative LDS byte offset the instruction wants.
__device__ __forceinline__ void async_to_lds_b128(void* lds_dst,
                                                  const void* gsrc) {
  uint32_t l = (uint32_t)(uintptr_t)lds_dst;
  asm volatile("global_load_async_to_lds_b128 %0, %1, off"
               :
               : "v"(l), "v"(gsrc)
               : "memory");
}

__device__ __forceinline__ void wait_async0() {
  asm volatile("s_wait_asynccnt 0" ::: "memory");
}

// ---------------------------- conversion kernels ---------------------------

// features (b, c, hw) f32  ->  (b, hw, c) bf16
__global__ __launch_bounds__(256) void k_feat_to_bf16(
    const float* __restrict__ in, __bf16* __restrict__ out, int total) {
  int i = blockIdx.x * blockDim.x + threadIdx.x;
  if (i >= total) return;
  int b  = i >> 21;              // / (2048*1024)
  int r  = i & ((1 << 21) - 1);
  int c  = r >> 10;
  int hw = r & 1023;
  out[((size_t)b * 1024 + hw) * 2048 + c] = (__bf16)in[i];
}

// weight (k, n) f32 -> transposed (n, k) bf16, so GEMM B-staging is contiguous
__global__ __launch_bounds__(256) void k_w_to_bf16_T(
    const float* __restrict__ in, __bf16* __restrict__ out, int total) {
  int i = blockIdx.x * blockDim.x + threadIdx.x;
  if (i >= total) return;
  int k = i >> 11;       // / 2048
  int n = i & 2047;
  out[(size_t)n * 2048 + k] = (__bf16)in[i];
}

// rel_emb (63, 256) f32 -> padded (64, 256) bf16 (row 63 zero)
__global__ __launch_bounds__(256) void k_rel_to_bf16(
    const float* __restrict__ in, __bf16* __restrict__ out) {
  int i = blockIdx.x * blockDim.x + threadIdx.x;
  if (i >= 64 * 256) return;
  out[i] = (i < 63 * 256) ? (__bf16)in[i] : (__bf16)0.f;
}

// copy raw features into d_out channels [0, 2048) of both halves
__global__ __launch_bounds__(256) void k_copy_passthrough(
    const float* __restrict__ l, const float* __restrict__ r,
    float* __restrict__ out, int total, size_t halfOff) {
  int i = blockIdx.x * blockDim.x + threadIdx.x;
  if (i >= total) return;
  int b  = i >> 21;
  int rm = i & ((1 << 21) - 1);
  int c  = rm >> 10;
  int hw = rm & 1023;
  size_t o = ((size_t)b * 4096 + c) * 1024 + hw;
  out[o] = l[i];
  out[halfOff + o] = r[i];
}

// ------------------------------- WMMA GEMM ---------------------------------
// C(4096x2048) = A(4096x2048,bf16) @ W + bias, W given TRANSPOSED: Wt[n][k].
// Block tile 128x128, BK=32, 256 threads = 8 waves, wave = 32x64 (2x4 WMMA).
// Double-buffered async LDS staging; prefetch 2 tiles ahead.
// mode 0: bf16 scatter into [b][head][w][h][c]        (Q/K layout)
// mode 1: f32 into d_out channels 2048+n (outBase = half offset), float4 x2
// mode 2: bf16 scatter into [b][head][w][c][h]        (V transposed layout)
__global__ __launch_bounds__(256) void k_gemm_bf16(
    const __bf16* __restrict__ A, const __bf16* __restrict__ Wt,
    const float* __restrict__ bias, void* __restrict__ out,
    int mode, size_t outBase) {
  constexpr int Kdim = 2048;
  __shared__ __bf16 As[2][128][40];  // [m][k], stride 40 halves (16B-aligned)
  __shared__ __bf16 Bs[2][128][40];  // [n][k]

  const int tid = threadIdx.x;
  const int m0 = blockIdx.y * 128;
  const int n0 = blockIdx.x * 128;

  const int wave  = tid >> 5, lane = tid & 31;
  const int lhalf = lane >> 4, lmod = lane & 15;
  const int wr = (wave >> 1) * 32;  // wave row offset in block tile
  const int wc = (wave & 1) * 64;   // wave col offset

  v8f acc[2][4];
  for (int i = 0; i < 2; ++i)
    for (int j = 0; j < 4; ++j) acc[i][j] = v8f_zero();

  const int sr = tid >> 1, sc = (tid & 1) * 16;  // staging: 16 bf16/thread/op
  const __bf16* aRow = A  + (size_t)(m0 + sr) * Kdim + sc;
  const __bf16* bRow = Wt + (size_t)(n0 + sr) * Kdim + sc;

  auto stage = [&](int buf, int kt) {
    async_to_lds_b128(&As[buf][sr][sc],     aRow + kt);
    async_to_lds_b128(&As[buf][sr][sc + 8], aRow + kt + 8);
    async_to_lds_b128(&Bs[buf][sr][sc],     bRow + kt);
    async_to_lds_b128(&Bs[buf][sr][sc + 8], bRow + kt + 8);
  };

  stage(0, 0);
  wait_async0();
  __syncthreads();

  int buf = 0;
  for (int kt = 0; kt < Kdim; kt += 32) {
    if (kt + 32 < Kdim) stage(buf ^ 1, kt + 32);  // async, overlaps WMMAs
    if (kt + 64 < Kdim) {
      __builtin_prefetch(aRow + kt + 64, 0, 1);   // global_prefetch_b8
      __builtin_prefetch(bRow + kt + 64, 0, 1);
    }

    BF16x16 af[2], bfr[4];
#pragma unroll
    for (int i = 0; i < 2; ++i) {
      const int m = wr + i * 16 + lmod;
      const int kb = lhalf * 8;   // A k-pattern: {kb..kb+7, kb+16..kb+23}
      af[i].q[0] = *(const uint4*)&As[buf][m][kb];
      af[i].q[1] = *(const uint4*)&As[buf][m][kb + 16];
    }
#pragma unroll
    for (int j = 0; j < 4; ++j) {
      const int n = wc + j * 16 + lmod;
      const int kb = lhalf * 8;   // B k-pattern: {kb..kb+7, kb+16..kb+23}
      bfr[j].q[0] = *(const uint4*)&Bs[buf][n][kb];
      bfr[j].q[1] = *(const uint4*)&Bs[buf][n][kb + 16];
    }
#pragma unroll
    for (int i = 0; i < 2; ++i)
#pragma unroll
      for (int j = 0; j < 4; ++j)
        acc[i][j] = __builtin_amdgcn_wmma_f32_16x16x32_bf16(
            false, af[i].v, false, bfr[j].v, (short)0, acc[i][j], false, false);

    wait_async0();    // next tile landed (issued before this tile's WMMAs)
    __syncthreads();
    buf ^= 1;
  }

  // epilogue: C/D layout row = v + 8*lhalf, col = lmod.
  // Consecutive v -> consecutive m -> consecutive w (and hw); h fixed per tile.
#pragma unroll
  for (int i = 0; i < 2; ++i) {
    const int m_base = m0 + wr + i * 16 + 8 * lhalf;  // m for v = 0
    const int b   = m_base >> 10;
    const int hw0 = m_base & 1023;
    const int h   = hw0 >> 5;
    const int w0  = hw0 & 31;
#pragma unroll
    for (int j = 0; j < 4; ++j) {
      const int n = n0 + wc + j * 16 + lmod;
      const float bsv = bias[n];
      if (mode == 1) {
        float* dst = (float*)out + outBase +
                     ((size_t)b * 4096 + 2048 + n) * 1024 + hw0;
        float4 f0 = {acc[i][j][0] + bsv, acc[i][j][1] + bsv,
                     acc[i][j][2] + bsv, acc[i][j][3] + bsv};
        float4 f1 = {acc[i][j][4] + bsv, acc[i][j][5] + bsv,
                     acc[i][j][6] + bsv, acc[i][j][7] + bsv};
        *(float4*)dst = f0;
        *(float4*)(dst + 4) = f1;
      } else {
        const int head = n >> 8, c = n & 255;
#pragma unroll
        for (int v = 0; v < 8; ++v) {
          const float val = acc[i][j][v] + bsv;
          const int w = w0 + v;
          size_t idx;
          if (mode == 0)   // [b][head][w][h][c]
            idx = ((((size_t)(b * 8 + head) * 32 + w) * 32 + h) * 256 + c);
          else             // mode 2: [b][head][w][c][h]
            idx = ((((size_t)(b * 8 + head) * 32 + w) * 256 + c) * 32 + h);
          ((__bf16*)out)[idx] = (__bf16)val;
        }
      }
    }
  }
}

// ------------------------------ attention core -----------------------------
// One block per (b, head, w): 32 queries (h) x 32 keys, head dim 256.
// Q,K arrive as [h][c] blocks; V arrives already transposed as [c][key].
// scores = Q K^T ; P = Q rel^T (32x64) ; scores += P[h,(k-h)%63]+P[h,(k-w)%63]
// probs = softmax(scores/16) ; att = probs V -> attOut[b*1024+h*32+w][head*256+c]
__global__ __launch_bounds__(128) void k_attention(
    const __bf16* __restrict__ Q, const __bf16* __restrict__ Kb,
    const __bf16* __restrict__ Vt_g, const __bf16* __restrict__ relp,
    __bf16* __restrict__ attOut) {
  __shared__ __bf16 Qs[32][264];   // row stride 528B: staggers banks, 16B-aligned
  __shared__ __bf16 Ks[32][264];
  __shared__ __bf16 Vt[256][40];   // [c][key]
  __shared__ float  Sc[32][33];
  __shared__ float  Pm[32][64];
  __shared__ __bf16 Pr[32][40];    // probs, padded for aligned A-frag loads

  const int w = blockIdx.x, head = blockIdx.y, b = blockIdx.z;
  const int tid = threadIdx.x;
  const int wave  = tid >> 5, lane = tid & 31;
  const int lhalf = lane >> 4, lmod = lane & 15;

  const size_t base = ((size_t)(b * 8 + head) * 32 + w) * 8192;
  const __bf16* Qg = Q    + base;
  const __bf16* Kg = Kb   + base;
  const __bf16* Vg = Vt_g + base;
  for (int i = tid; i < 1024; i += 128) {   // 8192 bf16 per operand, 16B units
    const int qr = i >> 5, qc = (i & 31) * 8;   // Q/K: [h][c]
    async_to_lds_b128(&Qs[qr][qc], Qg + i * 8);
    async_to_lds_b128(&Ks[qr][qc], Kg + i * 8);
    const int vc = i >> 2, vk = (i & 3) * 8;    // V: [c][key]
    async_to_lds_b128(&Vt[vc][vk], Vg + i * 8);
  }
  wait_async0();
  __syncthreads();

  // phase 1: scores (2x2 tiles, one per wave), K-loop over c
  {
    const int ti = wave >> 1, tj = wave & 1;
    v8f s = v8f_zero();
    for (int c0 = 0; c0 < 256; c0 += 32) {
      BF16x16 a, k2;
      const int m = ti * 16 + lmod;
      a.q[0] = *(const uint4*)&Qs[m][c0 + lhalf * 8];
      a.q[1] = *(const uint4*)&Qs[m][c0 + lhalf * 8 + 16];
      const int kk = tj * 16 + lmod;
      k2.q[0] = *(const uint4*)&Ks[kk][c0 + lhalf * 8];
      k2.q[1] = *(const uint4*)&Ks[kk][c0 + lhalf * 8 + 16];
      s = __builtin_amdgcn_wmma_f32_16x16x32_bf16(
          false, a.v, false, k2.v, (short)0, s, false, false);
    }
#pragma unroll
    for (int v = 0; v < 8; ++v)
      Sc[ti * 16 + v + 8 * lhalf][tj * 16 + lmod] = s[v];
  }

  // phase 2: P = Q @ rel^T (32x64); wave -> d-column block pj=wave, rows 0/1
  {
    const int pj = wave;
    v8f p0 = v8f_zero(), p1 = v8f_zero();
    for (int c0 = 0; c0 < 256; c0 += 32) {
      BF16x16 rb, a0, a1;
      const int d = pj * 16 + lmod;
      rb.q[0] = *(const uint4*)&relp[(size_t)d * 256 + c0 + lhalf * 8];
      rb.q[1] = *(const uint4*)&relp[(size_t)d * 256 + c0 + lhalf * 8 + 16];
      a0.q[0] = *(const uint4*)&Qs[lmod][c0 + lhalf * 8];
      a0.q[1] = *(const uint4*)&Qs[lmod][c0 + lhalf * 8 + 16];
      a1.q[0] = *(const uint4*)&Qs[16 + lmod][c0 + lhalf * 8];
      a1.q[1] = *(const uint4*)&Qs[16 + lmod][c0 + lhalf * 8 + 16];
      p0 = __builtin_amdgcn_wmma_f32_16x16x32_bf16(
          false, a0.v, false, rb.v, (short)0, p0, false, false);
      p1 = __builtin_amdgcn_wmma_f32_16x16x32_bf16(
          false, a1.v, false, rb.v, (short)0, p1, false, false);
    }
#pragma unroll
    for (int v = 0; v < 8; ++v) {
      Pm[v + 8 * lhalf][pj * 16 + lmod]      = p0[v];
      Pm[16 + v + 8 * lhalf][pj * 16 + lmod] = p1[v];
    }
  }
  __syncthreads();

  // phase 3: softmax rows (one thread per query row h)
  if (tid < 32) {
    const int h = tid;
    float buf[32], mx = -3.0e38f;
#pragma unroll
    for (int k = 0; k < 32; ++k) {
      float s = Sc[h][k] + Pm[h][(k - h + 63) % 63] + Pm[h][(k - w + 63) % 63];
      s *= 0.0625f;  // 1/sqrt(256)
      buf[k] = s;
      mx = fmaxf(mx, s);
    }
    float sum = 0.f;
#pragma unroll
    for (int k = 0; k < 32; ++k) { buf[k] = __expf(buf[k] - mx); sum += buf[k]; }
    const float inv = 1.f / sum;
#pragma unroll
    for (int k = 0; k < 32; ++k) Pr[h][k] = (__bf16)(buf[k] * inv);
  }
  __syncthreads();

  // phase 4: att = probs @ V (K=32 -> one WMMA per 16x16 tile)
  {
    BF16x16 pa[2];
#pragma unroll
    for (int oi = 0; oi < 2; ++oi) {
      const int m = oi * 16 + lmod;
      pa[oi].q[0] = *(const uint4*)&Pr[m][lhalf * 8];
      pa[oi].q[1] = *(const uint4*)&Pr[m][lhalf * 8 + 16];
    }
#pragma unroll
    for (int j = 0; j < 4; ++j) {
      const int oj = wave * 4 + j;
      BF16x16 vb;
      const int c = oj * 16 + lmod;
      vb.q[0] = *(const uint4*)&Vt[c][lhalf * 8];
      vb.q[1] = *(const uint4*)&Vt[c][lhalf * 8 + 16];
#pragma unroll
      for (int oi = 0; oi < 2; ++oi) {
        v8f o = __builtin_amdgcn_wmma_f32_16x16x32_bf16(
            false, pa[oi].v, false, vb.v, (short)0, v8f_zero(), false, false);
#pragma unroll
        for (int v = 0; v < 8; ++v) {
          const int hRow = oi * 16 + v + 8 * lhalf;
          const size_t r = (size_t)b * 1024 + (size_t)hRow * 32 + w;
          attOut[r * 2048 + head * 256 + oj * 16 + lmod] = (__bf16)o[v];
        }
      }
    }
  }
}

// ------------------------------- orchestration -----------------------------

extern "C" void kernel_launch(void* const* d_in, const int* in_sizes, int n_in,
                              void* d_out, int out_size, void* d_ws,
                              size_t ws_size, hipStream_t stream) {
  (void)in_sizes; (void)n_in; (void)out_size; (void)ws_size;

  const float* leftF  = (const float*)d_in[0];
  const float* rightF = (const float*)d_in[1];
  const float* Wq = (const float*)d_in[2];
  const float* bq = (const float*)d_in[3];
  const float* Wk = (const float*)d_in[4];
  const float* bk = (const float*)d_in[5];
  const float* Wv = (const float*)d_in[6];
  const float* bv = (const float*)d_in[7];
  const float* Wo = (const float*)d_in[8];
  const float* bo = (const float*)d_in[9];
  const float* rel = (const float*)d_in[10];

  constexpr size_t AB  = (size_t)4096 * 2048;        // feature matrix elems
  constexpr size_t WE  = (size_t)2048 * 2048;        // weight elems
  constexpr size_t QKV = (size_t)4 * 8 * 32 * 32 * 256;
  constexpr size_t HALF = (size_t)4 * 4096 * 1024;   // d_out half (elems)

  __bf16* ws   = (__bf16*)d_ws;                      // ~134 MB of scratch used
  __bf16* abfL = ws;
  __bf16* abfR = abfL + AB;
  __bf16* wqB  = abfR + AB;   // transposed [n][k]
  __bf16* wkB  = wqB + WE;
  __bf16* wvB  = wkB + WE;
  __bf16* woB  = wvB + WE;
  __bf16* relp = woB + WE;
  __bf16* qbuf = relp + 64 * 256;
  __bf16* kbuf = qbuf + QKV;
  __bf16* vbuf = kbuf + QKV;
  __bf16* attb = vbuf + QKV;

  const int featTot = (int)AB;   // 8,388,608
  const int wTot    = (int)WE;   // 4,194,304

  // conversions (weights transposed once; reused by both directions)
  k_feat_to_bf16<<<(featTot + 255) / 256, 256, 0, stream>>>(leftF, abfL, featTot);
  k_feat_to_bf16<<<(featTot + 255) / 256, 256, 0, stream>>>(rightF, abfR, featTot);
  k_w_to_bf16_T<<<(wTot + 255) / 256, 256, 0, stream>>>(Wq, wqB, wTot);
  k_w_to_bf16_T<<<(wTot + 255) / 256, 256, 0, stream>>>(Wk, wkB, wTot);
  k_w_to_bf16_T<<<(wTot + 255) / 256, 256, 0, stream>>>(Wv, wvB, wTot);
  k_w_to_bf16_T<<<(wTot + 255) / 256, 256, 0, stream>>>(Wo, woB, wTot);
  k_rel_to_bf16<<<(64 * 256 + 255) / 256, 256, 0, stream>>>(rel, relp);

  const dim3 gGrid(16, 32), gBlk(256);
  const dim3 aGrid(32, 8, 4), aBlk(128);

  // direction 1: q from left, k/v from right -> weighted_r -> left half
  k_gemm_bf16<<<gGrid, gBlk, 0, stream>>>(abfL, wqB, bq, qbuf, 0, 0);
  k_gemm_bf16<<<gGrid, gBlk, 0, stream>>>(abfR, wkB, bk, kbuf, 0, 0);
  k_gemm_bf16<<<gGrid, gBlk, 0, stream>>>(abfR, wvB, bv, vbuf, 2, 0);
  k_attention<<<aGrid, aBlk, 0, stream>>>(qbuf, kbuf, vbuf, relp, attb);
  k_gemm_bf16<<<gGrid, gBlk, 0, stream>>>(attb, woB, bo, d_out, 1, 0);

  // direction 2: q from right, k/v from left -> weighted_l -> right half
  k_gemm_bf16<<<gGrid, gBlk, 0, stream>>>(abfR, wqB, bq, qbuf, 0, 0);
  k_gemm_bf16<<<gGrid, gBlk, 0, stream>>>(abfL, wkB, bk, kbuf, 0, 0);
  k_gemm_bf16<<<gGrid, gBlk, 0, stream>>>(abfL, wvB, bv, vbuf, 2, 0);
  k_attention<<<aGrid, aBlk, 0, stream>>>(qbuf, kbuf, vbuf, relp, attb);
  k_gemm_bf16<<<gGrid, gBlk, 0, stream>>>(attb, woB, bo, d_out, 1, HALF);

  // passthrough channels [0,2048) of both halves
  k_copy_passthrough<<<(featTot + 255) / 256, 256, 0, stream>>>(
      leftF, rightF, (float*)d_out, featTot, HALF);
}
